// SimpleRNN_22368189677729
// MI455X (gfx1250) — matrix operand order repaired and checked by reading, not compiled, not run on hardware
//
#include <hip/hip_runtime.h>
#include <hip/hip_bf16.h>
#include <stdint.h>

// ---------------- problem dims (fixed by reference) ----------------
#define T_STEPS 512
#define BATCH   64
#define IN_DIM  512
#define HID     1024
#define OUT_DIM 512
#define ROWS    (T_STEPS * BATCH)      // 32768

typedef __attribute__((ext_vector_type(16))) __bf16 bf16x16;
typedef __attribute__((ext_vector_type(8)))  float  f32x8;

union Frag32B { uint4 u4[2]; bf16x16 v; };

__device__ __forceinline__ unsigned short f32_to_bf16(float f) {
  unsigned int u = __float_as_uint(f);
  u += 0x7FFFu + ((u >> 16) & 1u);      // round-to-nearest-even
  return (unsigned short)(u >> 16);
}

__device__ __forceinline__ f32x8 wmma_bf16(const bf16x16 a, const bf16x16 b,
                                           const f32x8 c) {
  return __builtin_amdgcn_wmma_f32_16x16x32_bf16(
      /*neg_a=*/false, a, /*neg_b=*/false, b,
      /*c_mod=*/(short)0, c, /*reuse_a=*/false, /*reuse_b=*/false);
}

// CDNA5 async memory->LDS copy (ASYNCcnt-tracked), 16B per lane.
// dst VGPR holds the wave-relative LDS byte address; addr is 64-bit global.
__device__ __forceinline__ void async_copy_b128(unsigned lds_off,
                                                const void* gptr) {
  asm volatile("global_load_async_to_lds_b128 %0, %1, off"
               :: "v"(lds_off), "v"(gptr)
               : "memory");
}

__device__ __forceinline__ void wait_async0() {
  asm volatile("s_wait_asynccnt 0x0" ::: "memory");
}

__device__ __forceinline__ unsigned lds_addr_of(const void* p) {
  // generic pointer to LDS: low 32 bits are the wave-relative LDS offset
  return (unsigned)(uintptr_t)p;
}

// ---------------- f32 -> bf16 conversion (grid-stride) ----------------
__global__ void convert_f32_to_bf16(const float* __restrict__ src,
                                    unsigned short* __restrict__ dst, int n) {
  int i = blockIdx.x * blockDim.x + threadIdx.x;
  int stride = gridDim.x * blockDim.x;
  for (; i < n; i += stride) dst[i] = f32_to_bf16(src[i]);
}

// ---------------- LDS-tiled bf16 GEMM:  C[M,N] = A[M,K] * B[N,K]^T + bias ---
// A, B bf16 row-major; C f32. Tile: 128(M) x 64(N), K-chunk 32.
// 256 threads = 8 waves, wave grid 4(M) x 2(N), 32x32 per wave (4 wmma frags).
// Double-buffered LDS fed by async-to-LDS copies: movement overlaps WMMA.
__global__ __launch_bounds__(256)
void gemm_bf16_tn(const unsigned short* __restrict__ A,
                  const unsigned short* __restrict__ B,
                  const float* __restrict__ bias,
                  float* __restrict__ C,
                  int M, int N, int K) {
  constexpr int MT = 128, NT = 64;
  constexpr int LDA = 40, LDB = 40;            // padded bf16 row stride (16B aligned)
  __shared__ unsigned short As[2][MT * LDA];   // 2 x 10 KB
  __shared__ unsigned short Bs[2][NT * LDB];   // 2 x 5 KB

  const int tid  = threadIdx.x;
  const int lane = tid & 31;
  const int wv   = tid >> 5;
  const int mw   = (wv & 3) * 32;              // wave M offset in tile
  const int nw   = (wv >> 2) * 32;             // wave N offset in tile
  const int m0   = blockIdx.y * MT;
  const int n0   = blockIdx.x * NT;
  const int l16  = lane & 15;
  const int half = lane >> 4;

  // staging: every thread moves 3 x 16B per K-chunk (A rows 0..63, 64..127, B)
  const int srow = tid >> 2;                   // 0..63
  const int sseg = tid & 3;                    // 4 segs of 8 bf16 (16B)

  unsigned ldsA0[2], ldsA1[2], ldsB0[2];
#pragma unroll
  for (int b = 0; b < 2; ++b) {
    ldsA0[b] = lds_addr_of(&As[b][srow * LDA + sseg * 8]);
    ldsA1[b] = lds_addr_of(&As[b][(64 + srow) * LDA + sseg * 8]);
    ldsB0[b] = lds_addr_of(&Bs[b][srow * LDB + sseg * 8]);
  }
  const unsigned short* gA0 = A + (size_t)(m0 + srow) * K + sseg * 8;
  const unsigned short* gA1 = A + (size_t)(m0 + 64 + srow) * K + sseg * 8;
  const unsigned short* gB0 = B + (size_t)(n0 + srow) * K + sseg * 8;

  auto stage = [&](int k0, int buf) {
    async_copy_b128(ldsA0[buf], gA0 + k0);
    async_copy_b128(ldsA1[buf], gA1 + k0);
    async_copy_b128(ldsB0[buf], gB0 + k0);
  };

  f32x8 acc[2][2] = {};
  const int nk = K / 32;
  stage(0, 0);

  for (int kc = 0; kc < nk; ++kc) {
    const int buf = kc & 1;
    wait_async0();        // this wave's copies into buf are done
    __syncthreads();      // everyone's copies done; prior readers of buf^1 done
    if (kc + 1 < nk) stage((kc + 1) * 32, buf ^ 1);

    const unsigned short* Asb = As[buf];
    const unsigned short* Bsb = Bs[buf];
    Frag32B a[2], b[2];
#pragma unroll
    for (int i = 0; i < 2; ++i) {
      const unsigned short* ap = Asb + (mw + i * 16 + l16) * LDA;
      a[i].u4[0] = *(const uint4*)(ap + 8 * half);        // K = 8h .. 8h+7
      a[i].u4[1] = *(const uint4*)(ap + 16 + 8 * half);   // K = 16+8h .. 23+8h
      const unsigned short* bp = Bsb + (nw + i * 16 + l16) * LDB;
      b[i].u4[0] = *(const uint4*)(bp + 8 * half);        // K = 8h .. 8h+7
      b[i].u4[1] = *(const uint4*)(bp + 16 + 8 * half);   // K = 16+8h .. 23+8h
    }
#pragma unroll
    for (int i = 0; i < 2; ++i)
#pragma unroll
      for (int j = 0; j < 2; ++j)
        acc[i][j] = wmma_bf16(a[i].v, b[j].v, acc[i][j]);
  }

  // epilogue: D frag element r -> row r + 8*half, col lane%16
#pragma unroll
  for (int i = 0; i < 2; ++i) {
    const int mrow = m0 + mw + i * 16 + half * 8;
#pragma unroll
    for (int j = 0; j < 2; ++j) {
      const int ncol = n0 + nw + j * 16 + l16;
      const float bn = bias[ncol];
      float* cp = C + (size_t)mrow * N + ncol;
#pragma unroll
      for (int r = 0; r < 8; ++r)
        cp[(size_t)r * N] = acc[i][j][r] + bn;
    }
  }
}

// ---------------- persistent recurrent kernel ----------------
// 16 workgroups; WG w owns hidden columns [w*64, w*64+64). Its 64x1024 bf16
// slice of W_hh stays resident in LDS (~129 KB of the 320 KB WGP LDS).
// Each step: h_new = tanh(xp[t] + h @ W_hh^T + b_hh); ping-pong h through a
// small global buffer; device-wide atomic barrier between steps.
__global__ __launch_bounds__(256)
void rnn_steps(const unsigned short* __restrict__ Whh,   // [HID, HID] bf16
               const float* __restrict__ xp,             // [T, B, H] f32
               const float* __restrict__ b_hh,           // [H]
               unsigned short* __restrict__ hbuf,        // [2, B, H] bf16
               unsigned short* __restrict__ hs,          // [T, B, H] bf16
               float* __restrict__ h_last,               // [B, H] f32
               unsigned int* __restrict__ bar) {
  extern __shared__ unsigned short Wl[];
  constexpr int LDW = 1032;                    // padded K stride (bank-friendly)
  const int tid  = threadIdx.x;
  const int lane = tid & 31, wv = tid >> 5;
  const int l16  = lane & 15, half = lane >> 4;
  const int nbase = blockIdx.x * 64;           // this WG's N slice
  const unsigned int ngrids = gridDim.x;

  // one-time: async-stage W_hh slice rows [nbase, nbase+64) into LDS
  {
    const int row = tid >> 2, seg = tid & 3;   // 64 rows x 4 segs of 256 elems
    const unsigned short* src = Whh + (size_t)(nbase + row) * HID + seg * 256;
    unsigned short* dst = Wl + row * LDW + seg * 256;
#pragma unroll
    for (int j = 0; j < 32; ++j)
      async_copy_b128(lds_addr_of(dst + j * 8), src + j * 8);
    wait_async0();
  }
  __syncthreads();

  const int mtile = (wv & 3) * 16;             // wave's M tile (batch rows)
  const int ng    = (wv >> 2) * 32;            // wave's local N offset (2 frags)
  const int am    = mtile + l16;               // A-frag row for this lane

  for (int t = 0; t < T_STEPS; ++t) {
    const unsigned short* hr = hbuf + (size_t)(t & 1) * (BATCH * HID);
    unsigned short*       hw = hbuf + (size_t)((t + 1) & 1) * (BATCH * HID);

    f32x8 acc[2] = {};
#pragma unroll 4
    for (int kc = 0; kc < HID / 32; ++kc) {
      const int k0 = kc * 32;
      Frag32B a, b0, b1;
      const unsigned short* ap = hr + (size_t)am * HID + k0 + 16 * half;
      a.u4[0] = *(const uint4*)(ap);
      a.u4[1] = *(const uint4*)(ap + 8);
      const unsigned short* bp0 = Wl + (ng + l16) * LDW + k0 + 16 * half;
      b0.u4[0] = *(const uint4*)(bp0);
      b0.u4[1] = *(const uint4*)(bp0 + 8);
      const unsigned short* bp1 = Wl + (ng + 16 + l16) * LDW + k0 + 16 * half;
      b1.u4[0] = *(const uint4*)(bp1);
      b1.u4[1] = *(const uint4*)(bp1 + 8);
      acc[0] = wmma_bf16(a.v, b0.v, acc[0]);
      acc[1] = wmma_bf16(a.v, b1.v, acc[1]);
    }

    // epilogue: add xp + b_hh, tanh, emit bf16 h and hs (f32 h_last at final t)
    const float* xpt = xp + (size_t)t * (BATCH * HID);
#pragma unroll
    for (int j = 0; j < 2; ++j) {
      const int n = nbase + ng + j * 16 + l16;
      const float bn = b_hh[n];
#pragma unroll
      for (int r = 0; r < 8; ++r) {
        const int m = mtile + half * 8 + r;
        const float v = acc[j][r] + xpt[m * HID + n] + bn;
        const float h = tanhf(v);
        const unsigned short hb = f32_to_bf16(h);
        hw[m * HID + n] = hb;
        hs[(size_t)t * (BATCH * HID) + m * HID + n] = hb;
        if (t == T_STEPS - 1) h_last[m * HID + n] = h;
      }
    }

    // device-wide arrive + spin barrier (monotone counter, reset per launch)
    __syncthreads();
    if (tid == 0) {
      __threadfence();  // release h writes to device scope
      __hip_atomic_fetch_add(bar, 1u, __ATOMIC_ACQ_REL, __HIP_MEMORY_SCOPE_AGENT);
      const unsigned int want = (unsigned int)(t + 1) * ngrids;
      while (__hip_atomic_load(bar, __ATOMIC_ACQUIRE, __HIP_MEMORY_SCOPE_AGENT) < want)
        __builtin_amdgcn_s_sleep(1);
    }
    __syncthreads();
    __threadfence();    // acquire for all lanes before reading peers' h
  }
}

// ---------------- host-side orchestration ----------------
extern "C" void kernel_launch(void* const* d_in, const int* in_sizes, int n_in,
                              void* d_out, int out_size, void* d_ws, size_t ws_size,
                              hipStream_t stream) {
  const float* x      = (const float*)d_in[0];   // [T,B,I]
  const float* hidden = (const float*)d_in[1];   // [1,B,H]
  const float* W_ih   = (const float*)d_in[2];   // [H,I]
  const float* W_hh   = (const float*)d_in[3];   // [H,H]
  const float* b_ih   = (const float*)d_in[4];   // [H]
  const float* b_hh   = (const float*)d_in[5];   // [H]
  const float* W_fc   = (const float*)d_in[6];   // [O,H]
  const float* b_fc   = (const float*)d_in[7];   // [O]
  float* out = (float*)d_out;                    // [T*B*O] ++ [B*H]

  // workspace carve-up (256B aligned regions)
  char* ws = (char*)d_ws;
  size_t off = 0;
  auto carve = [&](size_t bytes) -> char* {
    char* p = ws + off;
    off += (bytes + 255) & ~(size_t)255;
    return p;
  };
  unsigned short* xb   = (unsigned short*)carve((size_t)ROWS * IN_DIM * 2);
  unsigned short* wihb = (unsigned short*)carve((size_t)HID * IN_DIM * 2);
  unsigned short* whhb = (unsigned short*)carve((size_t)HID * HID * 2);
  unsigned short* wfcb = (unsigned short*)carve((size_t)OUT_DIM * HID * 2);
  unsigned short* hbuf = (unsigned short*)carve((size_t)2 * BATCH * HID * 2);
  float*          xpf  = (float*)carve((size_t)ROWS * HID * 4);
  unsigned short* hsb  = (unsigned short*)carve((size_t)ROWS * HID * 2);
  unsigned int*   bar  = (unsigned int*)carve(256);

  hipMemsetAsync(bar, 0, 256, stream);

  // bf16 conversions
  convert_f32_to_bf16<<<4096, 256, 0, stream>>>(x, xb, ROWS * IN_DIM);
  convert_f32_to_bf16<<<512, 256, 0, stream>>>(W_ih, wihb, HID * IN_DIM);
  convert_f32_to_bf16<<<1024, 256, 0, stream>>>(W_hh, whhb, HID * HID);
  convert_f32_to_bf16<<<512, 256, 0, stream>>>(W_fc, wfcb, OUT_DIM * HID);
  convert_f32_to_bf16<<<64, 256, 0, stream>>>(hidden, hbuf, BATCH * HID);

  // xp = x @ W_ih^T + b_ih   (M=32768, N=1024, K=512)
  gemm_bf16_tn<<<dim3(HID / 64, ROWS / 128), 256, 0, stream>>>(
      xb, wihb, b_ih, xpf, ROWS, HID, IN_DIM);

  // serial recurrence, 16 persistent WGs, W_hh slice resident in LDS
  rnn_steps<<<16, 256, 64 * 1032 * 2, stream>>>(
      whhb, xpf, b_hh, hbuf, hsb, out + (size_t)ROWS * OUT_DIM, bar);

  // out = hs @ W_fc^T + b_fc  (M=32768, N=512, K=1024)
  gemm_bf16_tn<<<dim3(OUT_DIM / 64, ROWS / 128), 256, 0, stream>>>(
      hsb, wfcb, b_fc, out, ROWS, OUT_DIM, HID);
}
